// SelfAttentionModule_70222715289719
// MI455X (gfx1250) — matrix operand order repaired and checked by reading, not compiled, not run on hardware
//
#include <hip/hip_runtime.h>

// ---------------------------------------------------------------------------
// Types for CDNA5 WMMA (gfx1250, wave32)
// ---------------------------------------------------------------------------
typedef _Float16 h16 __attribute__((ext_vector_type(16)));
typedef float    f8  __attribute__((ext_vector_type(8)));

struct alignas(16) H8V { _Float16 h[8]; };

#define WMMA_F16(A, B, C) \
    __builtin_amdgcn_wmma_f32_16x16x32_f16(false, (A), false, (B), (short)0, (C), false, false)

// Problem constants
#define HID 512
#define NH 8
#define HS 64
#define BB 8
#define SS 1024
#define PP 2047
#define PPAD 2048
#define BAND 1040   // 65 tiles of 16 covering the 1039-wide skew band

// ---------------------------------------------------------------------------
// Fragment loaders (layouts per CDNA5 ISA 7.12.2)
// A 16x32 f16: lane L -> row = mBase + (L&15); halves 0..7 -> K = kOff + (L>>4)*8 + i
//                                         halves 8..15 -> K = kOff + 16 + (L>>4)*8 + i
// B 32x16 f16: lane L -> col n = nBase + (L&15); halves 0..7 -> K = kBase + (L>>4)*8 + i
//                                         halves 8..15 -> K = kBase + 16 + (L>>4)*8 + i
// C/D 16x16 f32: lane L, reg r -> m = (L>>4)*8 + r, n = (L&15)
// ---------------------------------------------------------------------------
__device__ __forceinline__ h16 load_a_frag(const _Float16* base, int stride,
                                           int mBase, int kOff, int lane) {
    int row = mBase + (lane & 15);
    int kb  = (lane >> 4) * 8;
    const _Float16* p = base + (size_t)row * stride + kOff;
    H8V lo = *(const H8V*)(p + kb);
    H8V hi = *(const H8V*)(p + 16 + kb);
    h16 a;
#pragma unroll
    for (int i = 0; i < 8; ++i) { a[i] = lo.h[i]; a[i + 8] = hi.h[i]; }
    return a;
}

__device__ __forceinline__ h16 load_b_frag(const _Float16* base, int stride,
                                           int nBase, int kBase, int lane) {
    int n  = nBase + (lane & 15);
    int k0 = kBase + (lane >> 4) * 8;
    const _Float16* p = base + (size_t)n * stride + k0;
    H8V lo = *(const H8V*)(p);
    H8V hi = *(const H8V*)(p + 16);
    h16 b;
#pragma unroll
    for (int i = 0; i < 8; ++i) { b[i] = lo.h[i]; b[i + 8] = hi.h[i]; }
    return b;
}

// ---------------------------------------------------------------------------
// Weight transpose + f32->f16 convert: Wt[n*512+k] = W[k*512+n]
// ---------------------------------------------------------------------------
__global__ __launch_bounds__(256)
void wconv_kernel(const float* __restrict__ W, _Float16* __restrict__ Wt) {
    int idx = blockIdx.x * 256 + threadIdx.x;        // over 512*512
    int n = idx >> 9, k = idx & 511;
    Wt[idx] = (_Float16)W[(size_t)k * HID + n];
}

// position embeddings -> f16, zero-padded to 2048 rows
__global__ __launch_bounds__(256)
void posconv_kernel(const float* __restrict__ pe, _Float16* __restrict__ out) {
    int idx = blockIdx.x * 256 + threadIdx.x;        // over 2048*512
    int p = idx >> 9;
    out[idx] = (p < PP) ? (_Float16)pe[idx] : (_Float16)0.0f;
}

// ---------------------------------------------------------------------------
// LayerNorm over rows of [8192, 512] -> f16. One wave per row, 16 elems/lane.
// ---------------------------------------------------------------------------
__global__ __launch_bounds__(256)
void ln_kernel(const float* __restrict__ x, const float* __restrict__ gamma,
               const float* __restrict__ beta, _Float16* __restrict__ y) {
    int row  = blockIdx.x * 8 + (threadIdx.x >> 5);
    int lane = threadIdx.x & 31;
    const float4* p4 = (const float4*)(x + (size_t)row * HID + lane * 16);
    float v[16];
#pragma unroll
    for (int i = 0; i < 4; ++i) {
        float4 t = p4[i];
        v[4 * i + 0] = t.x; v[4 * i + 1] = t.y; v[4 * i + 2] = t.z; v[4 * i + 3] = t.w;
    }
    float sum = 0.f, sq = 0.f;
#pragma unroll
    for (int i = 0; i < 16; ++i) { sum += v[i]; sq += v[i] * v[i]; }
#pragma unroll
    for (int m = 16; m >= 1; m >>= 1) {
        sum += __shfl_xor(sum, m, 32);
        sq  += __shfl_xor(sq,  m, 32);
    }
    float mu  = sum * (1.0f / HID);
    float var = sq * (1.0f / HID) - mu * mu;
    float inv = rsqrtf(var + 1e-5f);
    _Float16* yr = y + (size_t)row * HID + lane * 16;
#pragma unroll
    for (int i = 0; i < 16; ++i) {
        int c = lane * 16 + i;
        yr[i] = (_Float16)((v[i] - mu) * inv * gamma[c] + beta[c]);
    }
}

// ---------------------------------------------------------------------------
// WMMA GEMM: C[M,512] = A[M,512] x W[512,512] (+bias). Wt is pre-transposed
// [n][k]. One wave computes a 16x64 tile. Software-pipelined k-loop: next
// iteration's fragments are issued before the current WMMAs consume theirs.
// ---------------------------------------------------------------------------
template <bool F32OUT>
__global__ __launch_bounds__(32)
void gemm_kernel(const _Float16* __restrict__ A, const _Float16* __restrict__ Wt,
                 const float* __restrict__ bias, void* __restrict__ out) {
    int lane   = threadIdx.x;
    int nChunk = blockIdx.x & 7;       // 512 / 64
    int mTile  = blockIdx.x >> 3;
    int m0 = mTile * 16, n0 = nChunk * 64;
    f8 acc0 = {}, acc1 = {}, acc2 = {}, acc3 = {};

    h16 a  = load_a_frag(A, HID, m0, 0, lane);
    h16 b0 = load_b_frag(Wt, HID, n0 + 0,  0, lane);
    h16 b1 = load_b_frag(Wt, HID, n0 + 16, 0, lane);
    h16 b2 = load_b_frag(Wt, HID, n0 + 32, 0, lane);
    h16 b3 = load_b_frag(Wt, HID, n0 + 48, 0, lane);

    for (int k0 = 0; k0 < HID; k0 += 32) {
        h16 na = a, nb0 = b0, nb1 = b1, nb2 = b2, nb3 = b3;
        int kn = k0 + 32;
        if (kn < HID) {                   // issue next-iter loads first
            na  = load_a_frag(A, HID, m0, kn, lane);
            nb0 = load_b_frag(Wt, HID, n0 + 0,  kn, lane);
            nb1 = load_b_frag(Wt, HID, n0 + 16, kn, lane);
            nb2 = load_b_frag(Wt, HID, n0 + 32, kn, lane);
            nb3 = load_b_frag(Wt, HID, n0 + 48, kn, lane);
        }
        acc0 = WMMA_F16(a, b0, acc0);
        acc1 = WMMA_F16(a, b1, acc1);
        acc2 = WMMA_F16(a, b2, acc2);
        acc3 = WMMA_F16(a, b3, acc3);
        a = na; b0 = nb0; b1 = nb1; b2 = nb2; b3 = nb3;
    }
    int mb = (lane >> 4) * 8, n = lane & 15;
#pragma unroll
    for (int r = 0; r < 8; ++r) {
        int row = m0 + mb + r;
        float vals[4] = { acc0[r], acc1[r], acc2[r], acc3[r] };
#pragma unroll
        for (int j = 0; j < 4; ++j) {
            int col = n0 + j * 16 + n;
            float v = vals[j] + (bias ? bias[col] : 0.0f);
            if (F32OUT) ((float*)out)[(size_t)row * HID + col] = v;
            else        ((_Float16*)out)[(size_t)row * HID + col] = (_Float16)v;
        }
    }
}

// ---------------------------------------------------------------------------
// Prep: reshape QKV GEMM outputs [b*s, h*64+d] into attention layouts,
// adding pos_bias_u / pos_bias_v to the two Q copies; V is transposed.
// ---------------------------------------------------------------------------
__global__ __launch_bounds__(256)
void prep_kernel(const _Float16* __restrict__ qg, const _Float16* __restrict__ kg,
                 const _Float16* __restrict__ vg, const float* __restrict__ pbu,
                 const float* __restrict__ pbv, _Float16* __restrict__ qu,
                 _Float16* __restrict__ qv, _Float16* __restrict__ kr,
                 _Float16* __restrict__ vT) {
    int idx = blockIdx.x * 256 + threadIdx.x;          // over 8192*512
    int c = idx & 511; int bs = idx >> 9;
    int s = bs & (SS - 1); int b = bs >> 10;
    int h = c >> 6, d = c & 63; int bh = b * NH + h;
    size_t dst = ((size_t)bh * SS + s) * HS + d;
    float q = (float)qg[idx];
    qu[dst] = (_Float16)(q + pbu[c]);
    qv[dst] = (_Float16)(q + pbv[c]);
    kr[dst] = kg[idx];
    vT[((size_t)bh * HS + d) * SS + s] = vg[idx];
}

// pos GEMM output [2048, 512] -> [H, 2048, 64]
__global__ __launch_bounds__(256)
void posprep_kernel(const _Float16* __restrict__ pg, _Float16* __restrict__ posr) {
    int idx = blockIdx.x * 256 + threadIdx.x;          // over 2048*512
    int c = idx & 511; int p = idx >> 9;
    int h = c >> 6, d = c & 63;
    posr[((size_t)h * PPAD + p) * HS + d] = pg[idx];
}

// ---------------------------------------------------------------------------
// Fused flash attention with Transformer-XL skew.
// One wave per (b, h, 16-row query tile). Phase 1: WMMA Q_v x pos^T over the
// 1040-wide skew band into LDS (f16), double-buffered. Phase 2: flash loop
// over 32-key tiles: K and V fragments are issued back-to-back so V arrives
// during the softmax VALU work; next K tile is prefetched into L2.
// ---------------------------------------------------------------------------
__global__ __launch_bounds__(32)
void attn_kernel(const _Float16* __restrict__ qu, const _Float16* __restrict__ qv,
                 const _Float16* __restrict__ kr, const _Float16* __restrict__ vT,
                 const _Float16* __restrict__ posr, _Float16* __restrict__ ctx) {
    __shared__ _Float16 bd[16 * BAND];
    __shared__ _Float16 pt[16 * 32];

    int lane = threadIdx.x;
    int blk  = blockIdx.x;                 // b*8*64 + h*64 + stile
    int st = blk & 63;
    int bh = blk >> 6;
    int h  = bh & (NH - 1);
    int b  = bh >> 3;
    int s0 = st * 16;

    const _Float16* quP  = qu   + (size_t)bh * SS * HS;
    const _Float16* qvP  = qv   + (size_t)bh * SS * HS;
    const _Float16* kP   = kr   + (size_t)bh * SS * HS;
    const _Float16* vP   = vT   + (size_t)bh * HS * SS;
    const _Float16* posP = posr + (size_t)h * PPAD * HS;

    // Q fragments (reused across the whole block)
    h16 qu0 = load_a_frag(quP, HS, s0, 0,  lane);
    h16 qu1 = load_a_frag(quP, HS, s0, 32, lane);
    h16 qv0 = load_a_frag(qvP, HS, s0, 0,  lane);
    h16 qv1 = load_a_frag(qvP, HS, s0, 32, lane);

    // ---- Phase 1: bd band = Q_v x pos^T, p in [1008-s0, 1008-s0+1039] ----
    int pBase = 1008 - s0;
    int mb = (lane >> 4) * 8, nn = lane & 15;

    h16 pb0 = load_b_frag(posP, HS, pBase, 0,  lane);
    h16 pb1 = load_b_frag(posP, HS, pBase, 32, lane);
    for (int ptile = 0; ptile < BAND / 16; ++ptile) {
        h16 npb0 = pb0, npb1 = pb1;
        if (ptile + 1 < BAND / 16) {       // issue next tile's loads first
            int n1 = pBase + (ptile + 1) * 16;
            npb0 = load_b_frag(posP, HS, n1, 0,  lane);
            npb1 = load_b_frag(posP, HS, n1, 32, lane);
        }
        f8 c = {};
        c = WMMA_F16(qv0, pb0, c);
        c = WMMA_F16(qv1, pb1, c);
#pragma unroll
        for (int r = 0; r < 8; ++r)
            bd[(mb + r) * BAND + ptile * 16 + nn] = (_Float16)c[r];
        pb0 = npb0; pb1 = npb1;
    }
    __syncthreads();

    // ---- Phase 2: flash loop over key tiles of 32 ----
    f8 o0 = {}, o1 = {}, o2 = {}, o3 = {};
    float rmax[8], rsum[8];
#pragma unroll
    for (int r = 0; r < 8; ++r) { rmax[r] = -3.0e30f; rsum[r] = 0.0f; }

    for (int t0 = 0; t0 < SS; t0 += 32) {
        // Issue all K and V fragment loads up front; V overlaps the softmax.
        h16 kb00 = load_b_frag(kP, HS, t0,      0,  lane);
        h16 kb01 = load_b_frag(kP, HS, t0,      32, lane);
        h16 kb10 = load_b_frag(kP, HS, t0 + 16, 0,  lane);
        h16 kb11 = load_b_frag(kP, HS, t0 + 16, 32, lane);
        h16 vb0 = load_b_frag(vP, SS, 0,  t0, lane);
        h16 vb1 = load_b_frag(vP, SS, 16, t0, lane);
        h16 vb2 = load_b_frag(vP, SS, 32, t0, lane);
        h16 vb3 = load_b_frag(vP, SS, 48, t0, lane);

        // Speculative L2 prefetch of the next key tile (gfx1250 prefetch path)
        if (t0 + 32 < SS) {
            __builtin_prefetch(kP + (size_t)(t0 + 32 + (lane & 15)) * HS, 0, 3);
            __builtin_prefetch(vP + (size_t)(lane & 15) * SS + t0 + 32, 0, 3);
        }

        f8 sA = {}, sB = {};
        sA = WMMA_F16(qu0, kb00, sA);
        sA = WMMA_F16(qu1, kb01, sA);
        sB = WMMA_F16(qu0, kb10, sB);
        sB = WMMA_F16(qu1, kb11, sB);

        float sv0[8], sv1[8];
#pragma unroll
        for (int r = 0; r < 8; ++r) {
            int m  = mb + r;
            int c0 = t0 + nn;
            int c1 = t0 + 16 + nn;
            // skew: scores_bd_skewed[s][t] = bd[s][t - s + S - 1]; band-local
            // column index = t - m + 15
            sv0[r] = 0.125f * (sA[r] + (float)bd[m * BAND + (c0 - m + 15)]);
            sv1[r] = 0.125f * (sB[r] + (float)bd[m * BAND + (c1 - m + 15)]);
        }

#pragma unroll
        for (int r = 0; r < 8; ++r) {
            float mx = fmaxf(sv0[r], sv1[r]);
#pragma unroll
            for (int msk = 8; msk >= 1; msk >>= 1)
                mx = fmaxf(mx, __shfl_xor(mx, msk, 32));
            float mNew  = fmaxf(rmax[r], mx);
            float alpha = __expf(rmax[r] - mNew);
            rmax[r] = mNew;
            float p0 = __expf(sv0[r] - mNew);
            float p1 = __expf(sv1[r] - mNew);
            sv0[r] = p0; sv1[r] = p1;
            float ps = p0 + p1;
#pragma unroll
            for (int msk = 8; msk >= 1; msk >>= 1)
                ps += __shfl_xor(ps, msk, 32);
            rsum[r] = rsum[r] * alpha + ps;
            o0[r] *= alpha; o1[r] *= alpha; o2[r] *= alpha; o3[r] *= alpha;
        }

        // C-layout -> A-layout conversion for P via LDS
#pragma unroll
        for (int r = 0; r < 8; ++r) {
            pt[(mb + r) * 32 + nn]      = (_Float16)sv0[r];
            pt[(mb + r) * 32 + 16 + nn] = (_Float16)sv1[r];
        }
        __syncthreads();
        h16 pfrag = load_a_frag(pt, 32, 0, 0, lane);
        o0 = WMMA_F16(pfrag, vb0, o0);
        o1 = WMMA_F16(pfrag, vb1, o1);
        o2 = WMMA_F16(pfrag, vb2, o2);
        o3 = WMMA_F16(pfrag, vb3, o3);
        __syncthreads();
    }

    // ---- Epilogue: normalize and scatter to ctx[b, s, h*64+d] ----
#pragma unroll
    for (int r = 0; r < 8; ++r) {
        float inv = 1.0f / rsum[r];
        int s = s0 + mb + r;
        size_t rowOff = ((size_t)b * SS + s) * HID + h * HS;
        ctx[rowOff + 0  + nn] = (_Float16)(o0[r] * inv);
        ctx[rowOff + 16 + nn] = (_Float16)(o1[r] * inv);
        ctx[rowOff + 32 + nn] = (_Float16)(o2[r] * inv);
        ctx[rowOff + 48 + nn] = (_Float16)(o3[r] * inv);
    }
}

// ---------------------------------------------------------------------------
// Host-side launcher
// ---------------------------------------------------------------------------
extern "C" void kernel_launch(void* const* d_in, const int* in_sizes, int n_in,
                              void* d_out, int out_size, void* d_ws, size_t ws_size,
                              hipStream_t stream) {
    const float* x    = (const float*)d_in[0];
    const float* pe   = (const float*)d_in[1];
    const float* g    = (const float*)d_in[2];
    const float* bet  = (const float*)d_in[3];
    const float* wq   = (const float*)d_in[4];
    const float* bq   = (const float*)d_in[5];
    const float* wk   = (const float*)d_in[6];
    const float* bk   = (const float*)d_in[7];
    const float* wv   = (const float*)d_in[8];
    const float* bv   = (const float*)d_in[9];
    const float* wpos = (const float*)d_in[10];
    const float* pbu  = (const float*)d_in[11];
    const float* pbv  = (const float*)d_in[12];
    const float* wout = (const float*)d_in[13];
    const float* bout = (const float*)d_in[14];
    float* out = (float*)d_out;

    // Workspace carve-up (f16 staging buffers, 256B aligned)
    char* cur = (char*)d_ws;
    auto alloc = [&](size_t elems) {
        _Float16* p = (_Float16*)cur;
        cur += (elems * sizeof(_Float16) + 255) & ~(size_t)255;
        return p;
    };
    const size_t W  = (size_t)HID * HID;       // 262144
    const size_t BS = (size_t)BB * SS;         // 8192
    _Float16* wqT   = alloc(W);
    _Float16* wkT   = alloc(W);
    _Float16* wvT   = alloc(W);
    _Float16* wposT = alloc(W);
    _Float16* woutT = alloc(W);
    _Float16* posInH = alloc((size_t)PPAD * HID);
    _Float16* xln   = alloc(BS * HID);
    _Float16* qg    = alloc(BS * HID);
    _Float16* kg    = alloc(BS * HID);
    _Float16* vg    = alloc(BS * HID);
    _Float16* pg    = alloc((size_t)PPAD * HID);
    _Float16* quB   = alloc(BS * HID);
    _Float16* qvB   = alloc(BS * HID);
    _Float16* krB   = alloc(BS * HID);
    _Float16* vTB   = alloc(BS * HID);
    _Float16* posr  = alloc((size_t)NH * PPAD * HS);
    _Float16* ctx   = alloc(BS * HID);

    // 1) weight transpose + f16 convert
    wconv_kernel<<<W / 256, 256, 0, stream>>>(wq, wqT);
    wconv_kernel<<<W / 256, 256, 0, stream>>>(wk, wkT);
    wconv_kernel<<<W / 256, 256, 0, stream>>>(wv, wvT);
    wconv_kernel<<<W / 256, 256, 0, stream>>>(wpos, wposT);
    wconv_kernel<<<W / 256, 256, 0, stream>>>(wout, woutT);
    posconv_kernel<<<(PPAD * HID) / 256, 256, 0, stream>>>(pe, posInH);

    // 2) LayerNorm -> f16
    ln_kernel<<<BS / 8, 256, 0, stream>>>(x, g, bet, xln);

    // 3) QKV + pos projections (WMMA GEMMs)
    gemm_kernel<false><<<(BS / 16) * 8, 32, 0, stream>>>(xln, wqT, bq, qg);
    gemm_kernel<false><<<(BS / 16) * 8, 32, 0, stream>>>(xln, wkT, bk, kg);
    gemm_kernel<false><<<(BS / 16) * 8, 32, 0, stream>>>(xln, wvT, bv, vg);
    gemm_kernel<false><<<(PPAD / 16) * 8, 32, 0, stream>>>(posInH, wposT, nullptr, pg);

    // 4) reshape + head biases
    prep_kernel<<<(BS * HID) / 256, 256, 0, stream>>>(qg, kg, vg, pbu, pbv,
                                                      quB, qvB, krB, vTB);
    posprep_kernel<<<(PPAD * HID) / 256, 256, 0, stream>>>(pg, posr);

    // 5) fused flash attention with skewed relative positions
    attn_kernel<<<BB * NH * (SS / 16), 32, 0, stream>>>(quB, qvB, krB, vTB, posr, ctx);

    // 6) output projection (f32 result)
    gemm_kernel<true><<<(BS / 16) * 8, 32, 0, stream>>>(ctx, woutT, bout, out);
}
